// GCNLayer_50096498540571
// MI455X (gfx1250) — compile-verified, matrix operand order checked by reading, not run on hardware
//
#include <hip/hip_runtime.h>

typedef __attribute__((ext_vector_type(16))) __bf16 bf16x16;
typedef __attribute__((ext_vector_type(8)))  __bf16 bf16x8;
typedef __attribute__((ext_vector_type(8)))  float  f32x8;
typedef __attribute__((ext_vector_type(4)))  float  f32x4;
typedef __attribute__((ext_vector_type(4)))  int    v4i;

typedef __attribute__((address_space(1))) v4i* gv4i_p;   // global (AS1) int4*
typedef __attribute__((address_space(3))) v4i* lv4i_p;   // LDS (AS3) int4*

#define N_NODES 8192
#define NFEAT   512

#define BM 128
#define BN 128
#define BK 32
#define LDT 40   // LDS row pitch in bf16 (80B: 16B-aligned, r*20 mod 64 distinct -> conflict-free frags)

#if __has_builtin(__builtin_amdgcn_global_load_async_to_lds_b128)
#define HAVE_ASYNC_LDS 1
#else
#define HAVE_ASYNC_LDS 0
#endif

__device__ __forceinline__ void wait_async_zero()
{
#if HAVE_ASYNC_LDS
#if __has_builtin(__builtin_amdgcn_s_wait_asynccnt)
    __builtin_amdgcn_s_wait_asynccnt(0);
#else
    asm volatile("s_wait_asynccnt 0x0" ::: "memory");
#endif
#endif
}

#if HAVE_ASYNC_LDS
__device__ __forceinline__ void async_copy16(const void* g, void* l)
{
    __builtin_amdgcn_global_load_async_to_lds_b128(
        (gv4i_p)(void*)g, (lv4i_p)l, 0, 0);
}
#endif

// ---------------- kernel 1: d[i] = rsqrt(rowsum(adj)+1) ----------------
__global__ __launch_bounds__(256) void k_degree(const float* __restrict__ adj,
                                                float* __restrict__ dvec)
{
    const int row = blockIdx.x;
    const f32x4* __restrict__ p = (const f32x4*)(adj + (size_t)row * N_NODES);
    float s = 0.f;
    for (int t = threadIdx.x; t < N_NODES / 4; t += 256) {
        f32x4 v = __builtin_nontemporal_load(&p[t]);
        s += v.x + v.y + v.z + v.w;
    }
    __shared__ float red[256];
    red[threadIdx.x] = s;
    __syncthreads();
    for (int off = 128; off > 0; off >>= 1) {
        if (threadIdx.x < off) red[threadIdx.x] += red[threadIdx.x + off];
        __syncthreads();
    }
    if (threadIdx.x == 0) dvec[row] = rsqrtf(red[0] + 1.0f);  // +1 = self loop; deg>0 always
}

// ---------------- kernel 2: fsT[c][j] = bf16(d[j] * feat[j][c]) ----------------
__global__ __launch_bounds__(256) void k_scaleT(const float* __restrict__ feat,
                                                const float* __restrict__ dvec,
                                                __bf16* __restrict__ fsT)
{
    __shared__ float tile[32][33];
    const int tx = threadIdx.x, ty = threadIdx.y;          // block (32,8)
    const int j0 = blockIdx.x * 32, c0 = blockIdx.y * 32;
#pragma unroll
    for (int i = 0; i < 4; ++i)
        tile[ty + i * 8][tx] = feat[(size_t)(j0 + ty + i * 8) * NFEAT + c0 + tx];
    __syncthreads();
    const float dj = dvec[j0 + tx];
#pragma unroll
    for (int i = 0; i < 4; ++i)
        fsT[(size_t)(c0 + ty + i * 8) * N_NODES + j0 + tx] = (__bf16)(dj * tile[tx][ty + i * 8]);
}

// ---------------- kernel 3: WT[n][k] = bf16(W[k][n]) ----------------
__global__ __launch_bounds__(256) void k_wT(const float* __restrict__ W,
                                            __bf16* __restrict__ WT)
{
    __shared__ float tile[32][33];
    const int tx = threadIdx.x, ty = threadIdx.y;          // block (32,8)
    const int k0 = blockIdx.x * 32, n0 = blockIdx.y * 32;
#pragma unroll
    for (int i = 0; i < 4; ++i)
        tile[ty + i * 8][tx] = W[(size_t)(k0 + ty + i * 8) * NFEAT + n0 + tx];
    __syncthreads();
#pragma unroll
    for (int i = 0; i < 4; ++i)
        WT[(size_t)(n0 + ty + i * 8) * NFEAT + k0 + tx] = (__bf16)tile[tx][ty + i * 8];
}

// Load one 16x32 bf16 WMMA fragment from an LDS tile (row pitch LDT).
// Lane L<16 : row L,   K = {0..7, 16..23};  lane L>=16 : row L-16, K = {8..15, 24..31}.
__device__ __forceinline__ bf16x16 load_frag(const __bf16* rowbase, int sk)
{
    bf16x8 lo = *(const bf16x8*)(rowbase + sk * 8);
    bf16x8 hi = *(const bf16x8*)(rowbase + 16 + sk * 8);
    bf16x16 r;
#pragma unroll
    for (int i = 0; i < 8; ++i) { r[i] = lo[i]; r[i + 8] = hi[i]; }
    return r;
}

// Stage one 128x32 bf16 tile from global (row pitch gpitch) into LDS (pitch LDT).
// 2 x 16B per thread; async global->LDS when available.
__device__ __forceinline__ void stage_bf16_tile(const __bf16* __restrict__ g, size_t gpitch,
                                                __bf16* l, int t)
{
#pragma unroll
    for (int i = 0; i < 2; ++i) {
        int q = t + i * 256;
        int r = q >> 2, c = (q & 3) * 8;
#if HAVE_ASYNC_LDS
        async_copy16(g + (size_t)r * gpitch + c, &l[r * LDT + c]);
#else
        *(f32x4*)&l[r * LDT + c] = *(const f32x4*)(g + (size_t)r * gpitch + c);
#endif
    }
}

// A tile (adj fp32 128x32): load to regs (4 x 16B / thread) then convert+store to LDS.
__device__ __forceinline__ void load_a_regs(const float* __restrict__ g, int k0, int t, f32x4 v[4])
{
#pragma unroll
    for (int i = 0; i < 4; ++i) {
        int f = t + i * 256;
        int r = f >> 3, c = (f & 7) * 4;
        v[i] = *(const f32x4*)(g + (size_t)r * N_NODES + k0 + c);
    }
}

__device__ __forceinline__ void store_a_lds(__bf16* l, int t, const f32x4 v[4])
{
#pragma unroll
    for (int i = 0; i < 4; ++i) {
        int f = t + i * 256;
        int r = f >> 3, c = (f & 7) * 4;
        __bf16* dst = &l[r * LDT + c];
        dst[0] = (__bf16)v[i].x; dst[1] = (__bf16)v[i].y;
        dst[2] = (__bf16)v[i].z; dst[3] = (__bf16)v[i].w;
    }
}

// 8 WMMAs on the current LDS buffers (wave tile 32M x 64N).
__device__ __forceinline__ void compute_tile(const __bf16* As, const __bf16* Bs,
                                             int wm, int wn, int r16, int sk,
                                             f32x8 acc[2][4])
{
    bf16x16 afr[2], bfr[4];
#pragma unroll
    for (int tm = 0; tm < 2; ++tm)
        afr[tm] = load_frag(&As[(wm * 32 + tm * 16 + r16) * LDT], sk);
#pragma unroll
    for (int tn = 0; tn < 4; ++tn)
        bfr[tn] = load_frag(&Bs[(wn * 64 + tn * 16 + r16) * LDT], sk);
#pragma unroll
    for (int tm = 0; tm < 2; ++tm)
#pragma unroll
        for (int tn = 0; tn < 4; ++tn)
            acc[tm][tn] = __builtin_amdgcn_wmma_f32_16x16x32_bf16(
                false, afr[tm], false, bfr[tn], (short)0, acc[tm][tn], false, false);
}

// ---------------- kernel 4: h = d ⊙ (adj @ fs + fs)   (bf16 WMMA, f32 accum) ----------------
__global__ __launch_bounds__(256) void k_gemm_adj(const float* __restrict__ adj,
                                                  const __bf16* __restrict__ fsT,
                                                  const float* __restrict__ feat,
                                                  const float* __restrict__ dvec,
                                                  __bf16* __restrict__ h)
{
    __shared__ __align__(16) __bf16 As0[BM * LDT], As1[BM * LDT];
    __shared__ __align__(16) __bf16 Bs0[BN * LDT], Bs1[BN * LDT];

    const int t    = threadIdx.x;
    const int bn   = blockIdx.x;          // fast dim: 4 N-siblings co-resident -> share adj via L2
    const int bm   = blockIdx.y;
    const int wave = t >> 5, lane = t & 31;
    const int wm   = wave >> 1, wn = wave & 1;   // 4(M) x 2(N) waves; wave tile 32M x 64N
    const int r16  = lane & 15, sk = lane >> 4;

    const float*  arow = adj + (size_t)(bm * BM) * N_NODES;
    const __bf16* brow = fsT + (size_t)(bn * BN) * N_NODES;

    f32x8 acc[2][4] = {};
    __bf16 *Ac = As0, *An = As1, *Bc = Bs0, *Bn = Bs1;

    // prologue: fill buffer 0
    f32x4 av[4];
    load_a_regs(arow, 0, t, av);
    stage_bf16_tile(brow, N_NODES, Bc, t);
    store_a_lds(Ac, t, av);
    wait_async_zero();
    __syncthreads();

    for (int k0 = BK; k0 < N_NODES; k0 += BK) {
        load_a_regs(arow, k0, t, av);               // next A -> regs (overlaps compute)
        stage_bf16_tile(brow + k0, N_NODES, Bn, t); // next B -> LDS (async)
        compute_tile(Ac, Bc, wm, wn, r16, sk, acc);
        store_a_lds(An, t, av);                     // next A -> LDS (bf16)
        wait_async_zero();
        __syncthreads();
        __bf16* tp;
        tp = Ac; Ac = An; An = tp;
        tp = Bc; Bc = Bn; Bn = tp;
    }
    compute_tile(Ac, Bc, wm, wn, r16, sk, acc);

    // epilogue: h[i][c] = d[i] * (acc + d[i]*feat[i][c])   (identity-term fusion)
#pragma unroll
    for (int tm = 0; tm < 2; ++tm)
#pragma unroll
        for (int tn = 0; tn < 4; ++tn) {
            const int gc = bn * BN + wn * 64 + tn * 16 + r16;
#pragma unroll
            for (int v = 0; v < 8; ++v) {
                const int gr = bm * BM + wm * 32 + tm * 16 + v + sk * 8;
                const float dd = dvec[gr];
                const float val = dd * (acc[tm][tn][v] + dd * feat[(size_t)gr * NFEAT + gc]);
                h[(size_t)gr * NFEAT + gc] = (__bf16)val;
            }
        }
}

// ---------------- kernel 5: out = h @ W   (bf16 WMMA, f32 out) ----------------
__global__ __launch_bounds__(256) void k_gemm_w(const __bf16* __restrict__ h,
                                                const __bf16* __restrict__ WT,
                                                float* __restrict__ out)
{
    __shared__ __align__(16) __bf16 As0[BM * LDT], As1[BM * LDT];
    __shared__ __align__(16) __bf16 Bs0[BN * LDT], Bs1[BN * LDT];

    const int t    = threadIdx.x;
    const int bn   = blockIdx.x;
    const int bm   = blockIdx.y;
    const int wave = t >> 5, lane = t & 31;
    const int wm   = wave >> 1, wn = wave & 1;
    const int r16  = lane & 15, sk = lane >> 4;

    const __bf16* hrow = h  + (size_t)(bm * BM) * NFEAT;
    const __bf16* wrow = WT + (size_t)(bn * BN) * NFEAT;

    f32x8 acc[2][4] = {};
    __bf16 *Ac = As0, *An = As1, *Bc = Bs0, *Bn = Bs1;

    stage_bf16_tile(hrow, NFEAT, Ac, t);
    stage_bf16_tile(wrow, NFEAT, Bc, t);
    wait_async_zero();
    __syncthreads();

    for (int k0 = BK; k0 < NFEAT; k0 += BK) {
        stage_bf16_tile(hrow + k0, NFEAT, An, t);
        stage_bf16_tile(wrow + k0, NFEAT, Bn, t);
        compute_tile(Ac, Bc, wm, wn, r16, sk, acc);
        wait_async_zero();
        __syncthreads();
        __bf16* tp;
        tp = Ac; Ac = An; An = tp;
        tp = Bc; Bc = Bn; Bn = tp;
    }
    compute_tile(Ac, Bc, wm, wn, r16, sk, acc);

#pragma unroll
    for (int tm = 0; tm < 2; ++tm)
#pragma unroll
        for (int tn = 0; tn < 4; ++tn) {
            const int gc = bn * BN + wn * 64 + tn * 16 + r16;
#pragma unroll
            for (int v = 0; v < 8; ++v) {
                const int gr = bm * BM + wm * 32 + tm * 16 + v + sk * 8;
                out[(size_t)gr * NFEAT + gc] = acc[tm][tn][v];
            }
        }
}

extern "C" void kernel_launch(void* const* d_in, const int* in_sizes, int n_in,
                              void* d_out, int out_size, void* d_ws, size_t ws_size,
                              hipStream_t stream) {
    const float* adj  = (const float*)d_in[0];   // [8192 x 8192] f32
    const float* feat = (const float*)d_in[1];   // [8192 x 512]  f32
    const float* W    = (const float*)d_in[2];   // [512 x 512]   f32
    float*       out  = (float*)d_out;           // [8192 x 512]  f32

    char* ws = (char*)d_ws;
    float*  dvec = (float*)ws;                                    //  32 KB
    __bf16* fsT  = (__bf16*)(ws + 32768);                         //   8 MB  [512][8192]
    __bf16* h    = (__bf16*)(ws + 32768 + 8388608);               //   8 MB  [8192][512]
    __bf16* WT   = (__bf16*)(ws + 32768 + 2 * (size_t)8388608);   // 512 KB  [512][512]

    k_degree  <<<dim3(N_NODES), dim3(256), 0, stream>>>(adj, dvec);
    k_scaleT  <<<dim3(N_NODES / 32, NFEAT / 32), dim3(32, 8), 0, stream>>>(feat, dvec, fsT);
    k_wT      <<<dim3(NFEAT / 32, NFEAT / 32), dim3(32, 8), 0, stream>>>(W, WT);
    k_gemm_adj<<<dim3(NFEAT / BN, N_NODES / BM), dim3(256), 0, stream>>>(adj, fsT, feat, dvec, h);
    k_gemm_w  <<<dim3(NFEAT / BN, N_NODES / BM), dim3(256), 0, stream>>>(h, WT, out);
}
